// MemoryModule_18236431138887
// MI455X (gfx1250) — matrix operand order repaired and checked by reading, not compiled, not run on hardware
//
#include <hip/hip_runtime.h>

typedef _Float16 v16h __attribute__((ext_vector_type(16)));
typedef _Float16 v8h  __attribute__((ext_vector_type(8)));
typedef float    v8f  __attribute__((ext_vector_type(8)));

#define N_ROWS    262144
#define MEMD      512
#define FEA       128
#define BM        64              // rows of X per workgroup
#define SH_STRIDE 520             // f16 row stride in LDS (260 dw % 64 == 4 -> conflict-free)
#define LDS_BYTES (BM * SH_STRIDE * 2)   // 66,560 B -> up to 4 WGs/WGP

static_assert(LDS_BYTES <= 320 * 1024, "LDS budget");

// ---------------------------------------------------------------------------
// Prep kernel: W (512x128 f32) -> global f16 copies:
//   wrm[512][128] row-major   (phase-1 B operands: fea-contiguous per slot)
//   wt [128][512] transposed  (phase-3 B operands: slot-contiguous per feature)
// Both are 128 KB -> L2-resident for the whole main kernel.
// ---------------------------------------------------------------------------
__global__ __launch_bounds__(256)
void prep_w_kernel(const float* __restrict__ w,
                   _Float16* __restrict__ wrm,
                   _Float16* __restrict__ wt) {
    int i = blockIdx.x * 1024 + threadIdx.x;
    #pragma unroll
    for (int u = 0; u < 4; ++u, i += 256) {
        const float v = w[i];
        const int s = i >> 7;      // memory slot
        const int f = i & 127;     // feature
        wrm[i] = (_Float16)v;
        wt[f * MEMD + s] = (_Float16)v;
    }
}

__global__ __launch_bounds__(256)
void memmod_main_kernel(const float* __restrict__ x,
                        const _Float16* __restrict__ wrm,
                        const _Float16* __restrict__ wt,
                        float* __restrict__ out) {
    extern __shared__ char smem[];
    _Float16* sS = (_Float16*)smem;      // scores (f16) then probs (f16), [BM][SH_STRIDE]

    const int tid   = threadIdx.x;
    const int lane  = tid & 31;
    const int wv    = tid >> 5;          // wave id 0..7
    const int l16   = lane & 15;
    const int hi    = lane >> 4;         // lane half: 0 or 1
    const int off8  = hi * 8;            // A-fragment K offset
    const int off16 = hi * 16;           // unused (B fragments use off8 too)
    const int rowBase = blockIdx.x * BM;

    const int m0   = (wv & 3) * 16;      // row tile within block
    const int half = wv >> 2;            // 0 -> slots/features 0..; 1 -> upper half

    // ---------------- Phase 1: S = X * W^T  (per-wave 16 rows x 256 slots) -----
    // A fragments: x rows, f32 -> f16 (the only unavoidable conversions).
    const float* xrow = x + (size_t)(rowBase + m0 + l16) * FEA;
    v16h afrag[4];
    #pragma unroll
    for (int c = 0; c < 4; ++c) {
        const int k0 = 32 * c + off8;
        const float4 a = *(const float4*)(xrow + k0);
        const float4 b = *(const float4*)(xrow + k0 + 4);
        const float4 cc = *(const float4*)(xrow + k0 + 16);
        const float4 d = *(const float4*)(xrow + k0 + 20);
        v16h t;
        t[0]  = (_Float16)a.x;  t[1]  = (_Float16)a.y;  t[2]  = (_Float16)a.z;  t[3]  = (_Float16)a.w;
        t[4]  = (_Float16)b.x;  t[5]  = (_Float16)b.y;  t[6]  = (_Float16)b.z;  t[7]  = (_Float16)b.w;
        t[8]  = (_Float16)cc.x; t[9]  = (_Float16)cc.y; t[10] = (_Float16)cc.z; t[11] = (_Float16)cc.w;
        t[12] = (_Float16)d.x;  t[13] = (_Float16)d.y;  t[14] = (_Float16)d.z;  t[15] = (_Float16)d.w;
        afrag[c] = t;
    }

    for (int nt = 0; nt < 16; ++nt) {
        const int slot0 = half * 256 + nt * 16;
        const _Float16* wrow = wrm + (size_t)(slot0 + l16) * FEA;
        v8f acc = {0.f, 0.f, 0.f, 0.f, 0.f, 0.f, 0.f, 0.f};
        #pragma unroll
        for (int c = 0; c < 4; ++c) {
            const int k0 = 32 * c + off8;
            const v8h blo = *(const v8h*)(wrow + k0);
            const v8h bhi = *(const v8h*)(wrow + k0 + 16);
            const v16h bfr = __builtin_shufflevector(blo, bhi,
                              0, 1, 2, 3, 4, 5, 6, 7, 8, 9, 10, 11, 12, 13, 14, 15);
            acc = __builtin_amdgcn_wmma_f32_16x16x32_f16(
                false, afrag[c], false, bfr, (short)0, acc, false, false);
        }
        #pragma unroll
        for (int v = 0; v < 8; ++v)
            sS[(m0 + v + off8) * SH_STRIDE + slot0 + l16] = (_Float16)acc[v];
    }
    __syncthreads();

    // ---------------- Phase 2: row softmax on f16 scores, in-place f16 probs ---
    // Lane owns elements 8*lane + 256*j (j=0,1): 2x ds_load_b128, perfect banks.
    for (int rr = 0; rr < 8; ++rr) {
        const int r = wv * 8 + rr;
        _Float16* srow = sS + (size_t)r * SH_STRIDE;
        const v8h h0 = *(const v8h*)(srow + 8 * lane);
        const v8h h1 = *(const v8h*)(srow + 8 * lane + 256);
        float f[16];
        float mx = -3.40282347e38f;
        #pragma unroll
        for (int e = 0; e < 8; ++e) { f[e] = (float)h0[e]; f[8 + e] = (float)h1[e]; }
        #pragma unroll
        for (int e = 0; e < 16; ++e) mx = fmaxf(mx, f[e]);
        #pragma unroll
        for (int o = 16; o > 0; o >>= 1)
            mx = fmaxf(mx, __shfl_xor(mx, o, 32));
        float sum = 0.f;
        #pragma unroll
        for (int e = 0; e < 16; ++e) { f[e] = __expf(f[e] - mx); sum += f[e]; }
        #pragma unroll
        for (int o = 16; o > 0; o >>= 1)
            sum += __shfl_xor(sum, o, 32);
        const float inv = 1.f / sum;
        v8h p0, p1;
        #pragma unroll
        for (int e = 0; e < 8; ++e) {
            p0[e] = (_Float16)(f[e] * inv);
            p1[e] = (_Float16)(f[8 + e] * inv);
        }
        *(v8h*)(srow + 8 * lane)       = p0;
        *(v8h*)(srow + 8 * lane + 256) = p1;
    }
    __syncthreads();

    // ---------------- Phase 3: O = P * W  (per-wave 16 rows x 64 features) -----
    const int c0 = half * 64;            // feature-column base for this wave
    v8f acc3[4];
    #pragma unroll
    for (int nt = 0; nt < 4; ++nt)
        #pragma unroll
        for (int v = 0; v < 8; ++v)
            acc3[nt][v] = 0.f;

    const _Float16* prow = sS + (size_t)(m0 + l16) * SH_STRIDE;
    for (int kt = 0; kt < 16; ++kt) {
        const int k0 = 32 * kt;
        const v8h alo = *(const v8h*)(prow + k0 + off8);
        const v8h ahi = *(const v8h*)(prow + k0 + 16 + off8);
        const v16h afr = __builtin_shufflevector(alo, ahi,
                          0, 1, 2, 3, 4, 5, 6, 7, 8, 9, 10, 11, 12, 13, 14, 15);
        #pragma unroll
        for (int nt = 0; nt < 4; ++nt) {
            const _Float16* wtrow = wt + (size_t)(c0 + nt * 16 + l16) * MEMD + k0 + off8;
            const v8h blo = *(const v8h*)(wtrow);
            const v8h bhi = *(const v8h*)(wtrow + 16);
            const v16h bfr = __builtin_shufflevector(blo, bhi,
                              0, 1, 2, 3, 4, 5, 6, 7, 8, 9, 10, 11, 12, 13, 14, 15);
            acc3[nt] = __builtin_amdgcn_wmma_f32_16x16x32_f16(
                false, afr, false, bfr, (short)0, acc3[nt], false, false);
        }
    }

    // ---------------- Epilogue: store O (f32) ---------------------------------
    #pragma unroll
    for (int nt = 0; nt < 4; ++nt) {
        float* obase = out + (size_t)(rowBase + m0 + off8) * FEA + c0 + nt * 16 + l16;
        #pragma unroll
        for (int v = 0; v < 8; ++v)
            obase[(size_t)v * FEA] = acc3[nt][v];
    }
}

extern "C" void kernel_launch(void* const* d_in, const int* in_sizes, int n_in,
                              void* d_out, int out_size, void* d_ws, size_t ws_size,
                              hipStream_t stream) {
    (void)in_sizes; (void)n_in; (void)out_size; (void)ws_size;
    const float* x = (const float*)d_in[0];
    const float* w = (const float*)d_in[1];
    float* out = (float*)d_out;

    _Float16* wrm = (_Float16*)d_ws;                       // 512*128 f16 = 128 KB
    _Float16* wt  = (_Float16*)((char*)d_ws + MEMD * FEA * sizeof(_Float16));

    (void)hipFuncSetAttribute((const void*)memmod_main_kernel,
                              hipFuncAttributeMaxDynamicSharedMemorySize, LDS_BYTES);

    prep_w_kernel<<<dim3(MEMD * FEA / 1024), 256, 0, stream>>>(w, wrm, wt);
    memmod_main_kernel<<<dim3(N_ROWS / BM), 256, LDS_BYTES, stream>>>(x, wrm, wt, out);
}
